// STAttentionBlock2_51410758533173
// MI455X (gfx1250) — matrix operand order repaired and checked by reading, not compiled, not run on hardware
//
#include <hip/hip_runtime.h>

typedef __attribute__((ext_vector_type(16))) _Float16 v16h;
typedef __attribute__((ext_vector_type(8)))  float    v8f;

#define C_TOT  512
#define HEADS  8
#define CH     64
#define T_SZ   16
#define HW     576
#define L_S    576
#define NB_S   32
#define NB_T   1152
#define M_QKV  1536
#define SCL    0.3535533905932738f   /* 64^(-1/4) */

__device__ __forceinline__ v8f wmma16(v16h a, v16h b, v8f c) {
  // D = A(16x32 f16) * B(32x16 f16) + C(16x16 f32)
  return __builtin_amdgcn_wmma_f32_16x16x32_f16(false, a, false, b, (short)0, c, false, false);
}

// CDNA5 async global->LDS copy, 16 bytes per lane (tracked by ASYNCcnt).
__device__ __forceinline__ void async_b128(unsigned lds_off, const void* gaddr) {
  asm volatile("global_load_async_to_lds_b128 %0, %1, off"
               :: "v"(lds_off), "v"(gaddr) : "memory");
}
__device__ __forceinline__ void wait_async0() {
  asm volatile("s_wait_asynccnt 0x0" ::: "memory");
}

// ---------------------------------------------------------------------------
// one-time fp32 -> f16 weight conversion
// ---------------------------------------------------------------------------
__global__ void cvt_f16(const float* __restrict__ s, _Float16* __restrict__ d, int n) {
  int i = blockIdx.x * blockDim.x + threadIdx.x;
  if (i < n) d[i] = (_Float16)s[i];
}

// ---------------------------------------------------------------------------
// GroupNorm (spatial view): x[b,c,t,h,w] -> xn[bt][c][l] f16
// ---------------------------------------------------------------------------
__global__ void gn_spatial(const float* __restrict__ x, const float* __restrict__ gw,
                           const float* __restrict__ gb, _Float16* __restrict__ xn) {
  __shared__ float sm[18];
  int blk = blockIdx.x;
  int n = blk >> 5, g = blk & 31;
  int b = n >> 4, t = n & 15;
  int cb = g * 16;
  float s = 0.f, ss = 0.f;
  for (int i = threadIdx.x; i < 16 * L_S; i += blockDim.x) {
    int c = cb + i / L_S, l = i % L_S;
    float v = x[((size_t)(b * C_TOT + c) * T_SZ + t) * HW + l];
    s += v; ss += v * v;
  }
  int lane = threadIdx.x & 31, wid = threadIdx.x >> 5;
#pragma unroll
  for (int o = 16; o; o >>= 1) { s += __shfl_down(s, o, 32); ss += __shfl_down(ss, o, 32); }
  if (lane == 0) { sm[wid] = s; sm[8 + wid] = ss; }
  __syncthreads();
  if (threadIdx.x == 0) {
    float ts = 0.f, tss = 0.f;
    for (int i = 0; i < 8; ++i) { ts += sm[i]; tss += sm[8 + i]; }
    float mean = ts / (16.f * L_S);
    float var  = tss / (16.f * L_S) - mean * mean;
    sm[16] = mean; sm[17] = rsqrtf(var + 1e-5f);
  }
  __syncthreads();
  float mean = sm[16], rstd = sm[17];
  for (int i = threadIdx.x; i < 16 * L_S; i += blockDim.x) {
    int c = cb + i / L_S, l = i % L_S;
    float v = x[((size_t)(b * C_TOT + c) * T_SZ + t) * HW + l];
    float o = (v - mean) * rstd * gw[c] + gb[c];
    xn[((size_t)n * C_TOT + c) * L_S + l] = (_Float16)o;
  }
}

// ---------------------------------------------------------------------------
// GroupNorm (temporal view): x1[b,c,t,h,w] -> xn[nb][c][t] f16, nb=b*576+p
// ---------------------------------------------------------------------------
__global__ void gn_temporal(const float* __restrict__ x1, const float* __restrict__ gw,
                            const float* __restrict__ gb, _Float16* __restrict__ xn) {
  __shared__ float sm[18];
  int blk = blockIdx.x;
  int n = blk >> 5, g = blk & 31;
  int b = n / HW, p = n % HW;
  int cb = g * 16;
  int i = threadIdx.x;
  int c = cb + (i >> 4), t = i & 15;
  float v = x1[((size_t)(b * C_TOT + c) * T_SZ + t) * HW + p];
  float s = v, ss = v * v;
  int lane = threadIdx.x & 31, wid = threadIdx.x >> 5;
#pragma unroll
  for (int o = 16; o; o >>= 1) { s += __shfl_down(s, o, 32); ss += __shfl_down(ss, o, 32); }
  if (lane == 0) { sm[wid] = s; sm[8 + wid] = ss; }
  __syncthreads();
  if (threadIdx.x == 0) {
    float ts = 0.f, tss = 0.f;
    for (int k = 0; k < 8; ++k) { ts += sm[k]; tss += sm[8 + k]; }
    float mean = ts / 256.f;
    float var  = tss / 256.f - mean * mean;
    sm[16] = mean; sm[17] = rsqrtf(var + 1e-5f);
  }
  __syncthreads();
  float o = (v - sm[16]) * sm[17] * gw[c] + gb[c];
  xn[((size_t)n * C_TOT + c) * T_SZ + t] = (_Float16)o;
}

// ---------------------------------------------------------------------------
// LDS-tiled GEMM with async global->LDS double buffering.
//   C[m, n] = Wh[m, :] . Bm[nb][:, n]   (K = 512, f16 inputs, f32 accum)
// block = 128 (4 waves, one 16-row M-subtile each), N panel = 16*NSUB wide.
// MODE 0: Out f16 [nb][m][L] (+bias)      (QKV)
// MODE 1: Out f32 spatial scatter +res    (proj_s)
// MODE 2: Out f32 temporal scatter +res   (proj_t)
// ---------------------------------------------------------------------------
template <int NSUB, int MODE, int L>
__global__ void gemm_lds(const _Float16* __restrict__ Wh, const float* __restrict__ bias,
                         const _Float16* __restrict__ Bm, const float* __restrict__ res,
                         void* __restrict__ Out, int Mtot) {
  __shared__ _Float16 panel[2][2048];            // 2 x 4 KB (rows padded to 64 f16)
  const int n0 = blockIdx.x * (16 * NSUB);
  const int mt = blockIdx.y, nb = blockIdx.z;
  const int lane = threadIdx.x & 31, w = threadIdx.x >> 5;
  const int lo = lane & 15, half = lane >> 4;
  const size_t bbase = (size_t)nb * C_TOT * L;
  const int gm = mt * 64 + w * 16 + lo;
  const unsigned pbase = (unsigned)(size_t)(&panel[0][0]);
  const char* gB = (const char*)Bm;

  // issue async panel load for K-step k0 into buffer buf
  auto issue = [&](int buf, int k0) {
    const int nchunk = 64 * NSUB;                // 16-byte chunks (32 rows x 2*NSUB)
    for (int c = threadIdx.x; c < nchunk; c += 128) {
      int row = c / (2 * NSUB);
      int seg = c % (2 * NSUB);
      unsigned loff = pbase + (unsigned)buf * 4096u + (unsigned)row * 128u + (unsigned)seg * 16u;
      const char* g = gB + 2 * (bbase + (size_t)(k0 + row) * L + n0) + seg * 16;
      async_b128(loff, g);
    }
  };

  v8f acc[NSUB];
#pragma unroll
  for (int s = 0; s < NSUB; ++s) acc[s] = {};

  issue(0, 0);
  for (int k0 = 0; k0 < C_TOT; k0 += 32) {
    int buf = (k0 >> 5) & 1;
    wait_async0();
    __syncthreads();                 // publishes async writes; frees buf^1 for reuse
    if (k0 + 32 < C_TOT) issue(buf ^ 1, k0 + 32);

    v16h a;
#pragma unroll
    for (int j = 0; j < 8; ++j) {
      a[j]     = Wh[(size_t)gm * C_TOT + k0 + half * 16 + j];
      a[j + 8] = Wh[(size_t)gm * C_TOT + k0 + half * 16 + 8 + j];
    }
#pragma unroll
    for (int s = 0; s < NSUB; ++s) {
      v16h bv;
#pragma unroll
      for (int j = 0; j < 16; ++j)
        bv[j] = panel[buf][(half * 16 + j) * 64 + s * 16 + lo];
      acc[s] = wmma16(a, bv, acc[s]);
    }
  }

#pragma unroll
  for (int s = 0; s < NSUB; ++s) {
#pragma unroll
    for (int r = 0; r < 8; ++r) {
      int m = mt * 64 + w * 16 + r + 8 * half;
      int n = n0 + s * 16 + lo;
      if (MODE == 0) {
        ((_Float16*)Out)[(size_t)nb * Mtot * L + (size_t)m * L + n] = (_Float16)(acc[s][r] + bias[m]);
      } else if (MODE == 1) {
        int b = nb >> 4, t = nb & 15;
        size_t idx = ((size_t)(b * C_TOT + m) * T_SZ + t) * HW + n;
        ((float*)Out)[idx] = res[idx] + acc[s][r] + bias[m];
      } else {
        int b = nb / HW, p = nb % HW;
        size_t idx = ((size_t)(b * C_TOT + m) * T_SZ + n) * HW + p;
        ((float*)Out)[idx] = res[idx] + acc[s][r] + bias[m];
      }
    }
  }
}

// ---------------------------------------------------------------------------
// Spatial attention: one wave per (bt, head, 16-query block).
// ---------------------------------------------------------------------------
__global__ void attn_spatial(const _Float16* __restrict__ QKV, _Float16* __restrict__ AO) {
  __shared__ float sc[16][L_S];                 // 36 KB
  int tb = blockIdx.x, head = blockIdx.y, bt = blockIdx.z;
  int lane = threadIdx.x, lo = lane & 15, half = lane >> 4;
  size_t qb = (size_t)bt * M_QKV * L_S;

  v16h aq[2];
#pragma unroll
  for (int kk = 0; kk < 2; ++kk) {
#pragma unroll
    for (int j = 0; j < 8; ++j) {
      int c0 = kk * 32 + half * 16 + j;
      int c1 = kk * 32 + half * 16 + 8 + j;
      aq[kk][j]     = (_Float16)((float)QKV[qb + (size_t)(head * CH + c0) * L_S + tb * 16 + lo] * SCL);
      aq[kk][j + 8] = (_Float16)((float)QKV[qb + (size_t)(head * CH + c1) * L_S + tb * 16 + lo] * SCL);
    }
  }
  for (int st = 0; st < L_S / 16; ++st) {
    v8f acc = {};
#pragma unroll
    for (int kk = 0; kk < 2; ++kk) {
      v16h bk;
#pragma unroll
      for (int j = 0; j < 16; ++j) {
        int c = kk * 32 + half * 16 + j;
        bk[j] = (_Float16)((float)QKV[qb + (size_t)(C_TOT + head * CH + c) * L_S + st * 16 + lo] * SCL);
      }
      acc = wmma16(aq[kk], bk, acc);
    }
#pragma unroll
    for (int r = 0; r < 8; ++r) sc[r + 8 * half][st * 16 + lo] = acc[r];
  }
  __syncthreads();

  {
    int row = lo;
    float mx = -3.0e38f;
    for (int j = half; j < L_S; j += 2) mx = fmaxf(mx, sc[row][j]);
    mx = fmaxf(mx, __shfl_xor(mx, 16, 32));
    float sum = 0.f;
    for (int j = half; j < L_S; j += 2) { float e = __expf(sc[row][j] - mx); sc[row][j] = e; sum += e; }
    sum += __shfl_xor(sum, 16, 32);
    float inv = 1.f / sum;
    for (int j = half; j < L_S; j += 2) sc[row][j] *= inv;
  }
  __syncthreads();

  v8f oacc[4] = {};
  for (int k0 = 0; k0 < L_S; k0 += 32) {
    v16h aw;
#pragma unroll
    for (int j = 0; j < 8; ++j) {
      aw[j]     = (_Float16)sc[lo][k0 + half * 16 + j];
      aw[j + 8] = (_Float16)sc[lo][k0 + half * 16 + 8 + j];
    }
#pragma unroll
    for (int nt2 = 0; nt2 < 4; ++nt2) {
      v16h bv;
#pragma unroll
      for (int j = 0; j < 16; ++j)
        bv[j] = QKV[qb + (size_t)(2 * C_TOT + head * CH + nt2 * 16 + lo) * L_S + k0 + half * 16 + j];
      oacc[nt2] = wmma16(aw, bv, oacc[nt2]);
    }
  }
#pragma unroll
  for (int nt2 = 0; nt2 < 4; ++nt2)
#pragma unroll
    for (int r = 0; r < 8; ++r) {
      int tloc = r + 8 * half;
      AO[(size_t)bt * C_TOT * L_S + (size_t)(head * CH + nt2 * 16 + lo) * L_S + tb * 16 + tloc]
          = (_Float16)oacc[nt2][r];
    }
}

// ---------------------------------------------------------------------------
// Temporal attention: one wave per (head, nb). Causal + RPE on K and V.
// ---------------------------------------------------------------------------
__global__ void attn_temporal(const _Float16* __restrict__ QKV, const float* __restrict__ rpk,
                              const float* __restrict__ rpv, _Float16* __restrict__ AO) {
  __shared__ float wsm[16][16];
  int head = blockIdx.x, nb = blockIdx.y;
  int lane = threadIdx.x, lo = lane & 15, half = lane >> 4;
  size_t qb = (size_t)nb * M_QKV * T_SZ;

  v8f acc = {};
#pragma unroll
  for (int kk = 0; kk < 2; ++kk) {
    v16h aq, bk;
#pragma unroll
    for (int j = 0; j < 8; ++j) {
      int c0 = kk * 32 + half * 16 + j, c1 = kk * 32 + half * 16 + 8 + j;
      aq[j]     = (_Float16)((float)QKV[qb + (size_t)(head * CH + c0) * T_SZ + lo] * SCL);
      aq[j + 8] = (_Float16)((float)QKV[qb + (size_t)(head * CH + c1) * T_SZ + lo] * SCL);
    }
#pragma unroll
    for (int j = 0; j < 16; ++j) {
      int c = kk * 32 + half * 16 + j;
      bk[j] = (_Float16)((float)QKV[qb + (size_t)(C_TOT + head * CH + c) * T_SZ + lo] * SCL);
    }
    acc = wmma16(aq, bk, acc);
  }
  // RPE-K (reference's transposed add): w[t,s] += sum_c q[c,s]*scl * rpk[t-s+16][c]
  for (int c = 0; c < CH; ++c) {
    float qv = (float)QKV[qb + (size_t)(head * CH + c) * T_SZ + lo] * SCL;
#pragma unroll
    for (int r = 0; r < 8; ++r) {
      int t = r + 8 * half;
      acc[r] += qv * rpk[(t - lo + 16) * CH + c];
    }
  }
#pragma unroll
  for (int r = 0; r < 8; ++r) {
    int t = r + 8 * half;
    if (lo > t) acc[r] = -1e8f;
    float mx = acc[r];
#pragma unroll
    for (int m = 8; m; m >>= 1) mx = fmaxf(mx, __shfl_xor(mx, m, 32));
    float e = __expf(acc[r] - mx);
    float sum = e;
#pragma unroll
    for (int m = 8; m; m >>= 1) sum += __shfl_xor(sum, m, 32);
    acc[r] = e / sum;
    wsm[t][lo] = acc[r];
  }
  __syncthreads();

  v16h aw;
#pragma unroll
  for (int j = 0; j < 8; ++j) {
    aw[j]     = (half == 0) ? (_Float16)wsm[lo][j] : (_Float16)0.f;
    aw[j + 8] = (half == 0) ? (_Float16)wsm[lo][8 + j] : (_Float16)0.f;
  }
  v8f oacc[4] = {};
#pragma unroll
  for (int nt2 = 0; nt2 < 4; ++nt2) {
    v16h bv;
#pragma unroll
    for (int j = 0; j < 16; ++j)
      bv[j] = (half == 0)
            ? QKV[qb + (size_t)(2 * C_TOT + head * CH + nt2 * 16 + lo) * T_SZ + j]
            : (_Float16)0.f;
    oacc[nt2] = wmma16(aw, bv, oacc[nt2]);
#pragma unroll
    for (int r = 0; r < 8; ++r) {
      int t = r + 8 * half;
      int c = nt2 * 16 + lo;
      float add = 0.f;
#pragma unroll
      for (int s2 = 0; s2 < 16; ++s2) add += wsm[t][s2] * rpv[(t - s2 + 16) * CH + c];
      AO[(size_t)nb * C_TOT * T_SZ + (size_t)(head * CH + c) * T_SZ + t]
          = (_Float16)(oacc[nt2][r] + add);
    }
  }
}

// ---------------------------------------------------------------------------
extern "C" void kernel_launch(void* const* d_in, const int* in_sizes, int n_in,
                              void* d_out, int out_size, void* d_ws, size_t ws_size,
                              hipStream_t stream) {
  const float* x        = (const float*)d_in[0];
  const float* norm_s_w = (const float*)d_in[1];
  const float* norm_s_b = (const float*)d_in[2];
  const float* qkv_s_w  = (const float*)d_in[3];
  const float* qkv_s_b  = (const float*)d_in[4];
  const float* proj_s_w = (const float*)d_in[5];
  const float* proj_s_b = (const float*)d_in[6];
  const float* norm_t_w = (const float*)d_in[7];
  const float* norm_t_b = (const float*)d_in[8];
  const float* qkv_t_w  = (const float*)d_in[9];
  const float* qkv_t_b  = (const float*)d_in[10];
  const float* proj_t_w = (const float*)d_in[11];
  const float* proj_t_b = (const float*)d_in[12];
  const float* rpk      = (const float*)d_in[13];
  const float* rpv      = (const float*)d_in[14];
  float* out = (float*)d_out;

  const size_t NELT   = (size_t)2 * C_TOT * T_SZ * HW;      // 9,437,184
  const size_t SZ_X1  = NELT * sizeof(float);
  const size_t SZ_XN  = NELT * sizeof(_Float16);
  const size_t SZ_QKV = NELT * 3 * sizeof(_Float16);
  const size_t SZ_AO  = NELT * sizeof(_Float16);
  char* wsp = (char*)d_ws;
  float*    X1  = (float*)wsp;
  _Float16* XN  = (_Float16*)(wsp + SZ_X1);
  _Float16* QKV = (_Float16*)(wsp + SZ_X1 + SZ_XN);
  _Float16* AO  = (_Float16*)(wsp + SZ_X1 + SZ_XN + SZ_QKV);
  char* wh = wsp + SZ_X1 + SZ_XN + SZ_QKV + SZ_AO;
  _Float16* WqkvS = (_Float16*)wh;                                   // 1536x512
  _Float16* WprjS = (_Float16*)(wh + (size_t)M_QKV * C_TOT * 2);     // 512x512
  _Float16* WqkvT = (_Float16*)(wh + (size_t)(M_QKV + C_TOT) * C_TOT * 2);
  _Float16* WprjT = (_Float16*)(wh + (size_t)(2 * M_QKV + C_TOT) * C_TOT * 2);

  // one-time weight conversion to f16
  const int nwq = M_QKV * C_TOT, nwp = C_TOT * C_TOT;
  cvt_f16<<<(nwq + 255) / 256, 256, 0, stream>>>(qkv_s_w, WqkvS, nwq);
  cvt_f16<<<(nwp + 255) / 256, 256, 0, stream>>>(proj_s_w, WprjS, nwp);
  cvt_f16<<<(nwq + 255) / 256, 256, 0, stream>>>(qkv_t_w, WqkvT, nwq);
  cvt_f16<<<(nwp + 255) / 256, 256, 0, stream>>>(proj_t_w, WprjT, nwp);

  // ---- spatial stage ----
  gn_spatial<<<NB_S * 32, 256, 0, stream>>>(x, norm_s_w, norm_s_b, XN);
  gemm_lds<4, 0, L_S><<<dim3(L_S / 64, M_QKV / 64, NB_S), 128, 0, stream>>>(
      WqkvS, qkv_s_b, XN, nullptr, QKV, M_QKV);
  attn_spatial<<<dim3(L_S / 16, HEADS, NB_S), 32, 0, stream>>>(QKV, AO);
  gemm_lds<4, 1, L_S><<<dim3(L_S / 64, C_TOT / 64, NB_S), 128, 0, stream>>>(
      WprjS, proj_s_b, AO, x, X1, C_TOT);

  // ---- temporal stage ----
  gn_temporal<<<NB_T * 32, 256, 0, stream>>>(X1, norm_t_w, norm_t_b, XN);
  gemm_lds<1, 0, T_SZ><<<dim3(1, M_QKV / 64, NB_T), 128, 0, stream>>>(
      WqkvT, qkv_t_b, XN, nullptr, QKV, M_QKV);
  attn_temporal<<<dim3(HEADS, NB_T), 32, 0, stream>>>(QKV, rpk, rpv, AO);
  gemm_lds<1, 2, T_SZ><<<dim3(1, C_TOT / 64, NB_T), 128, 0, stream>>>(
      WprjT, proj_t_b, AO, X1, out, C_TOT);
}
